// Memory_sup_33389075759209
// MI455X (gfx1250) — matrix operand order repaired and checked by reading, not compiled, not run on hardware
//
#include <hip/hip_runtime.h>
#include <hip/hip_bf16.h>

// ---------------------------------------------------------------------------
// MI455X (gfx1250): pipeline as chained WMMA-f16 GEMMs with fused epilogues.
// Round 3: 3x3-conv input kept in a zero-padded [B][H][W+2][128] layout so
// the implicit-GEMM A fragments need no per-lane exec guards / zero fills;
// conv tap weights double-buffered in LDS via global_load_async_to_lds_b128.
// ---------------------------------------------------------------------------

typedef __attribute__((ext_vector_type(16))) _Float16 v16h;
typedef __attribute__((ext_vector_type(8)))  _Float16 v8h;
typedef __attribute__((ext_vector_type(8)))  float    v8f;

__device__ __forceinline__ v8f zero8f() {
  v8f r;
#pragma unroll
  for (int i = 0; i < 8; ++i) r[i] = 0.f;
  return r;
}

// Build a 16-element f16 fragment from two contiguous 8xf16 (16B) loads.
__device__ __forceinline__ v16h ld_pair(const _Float16* p0, const _Float16* p1) {
  v8h lo = *(const v8h*)p0;
  v8h hi = *(const v8h*)p1;
  v16h r;
#pragma unroll
  for (int i = 0; i < 8; ++i) { r[i] = lo[i]; r[i + 8] = hi[i]; }
  return r;
}

// A fragment (16x32, MxK) from row-major A[row, K], per CDNA5 ISA layout:
// lanes 0-15: M=lane, K = {0..7, 16..23}; lanes 16-31: M=lane-16, K = {8..15, 24..31}
__device__ __forceinline__ v16h ld_a(const _Float16* __restrict__ A, long lda,
                                     int row0, int k0, int lane) {
  const _Float16* base = A + (long)(row0 + (lane & 15)) * lda + k0 + 8 * (lane >> 4);
  return ld_pair(base, base + 16);
}

// B fragment (32x16, KxN) where B = W^T, W row-major [N, K]:
// lane holds N=lane&15; half h=lane>>4 covers K = k0+8h..k0+8h+7, k0+16+8h..k0+16+8h+7
__device__ __forceinline__ v16h ld_b(const _Float16* W, long ldw,
                                     int col0, int k0, int lane) {
  const _Float16* base = W + (long)(col0 + (lane & 15)) * ldw + k0 + 8 * (lane >> 4);
  return ld_pair(base, base + 16);
}

// Async global->LDS copy of 16 bytes per lane (CDNA5 ASYNCcnt path).
__device__ __forceinline__ void async_copy_b128(unsigned lds_addr, const _Float16* gaddr) {
  asm volatile("global_load_async_to_lds_b128 %0, %1, off"
               :: "v"(lds_addr), "v"(gaddr)
               : "memory");
}
__device__ __forceinline__ void wait_asynccnt0() {
  asm volatile("s_wait_asynccnt 0x0" ::: "memory");
}

// ------------------------------- weight prep -------------------------------

__global__ void k_cvt_f16(_Float16* __restrict__ dst, const float* __restrict__ src, int n) {
  int i = blockIdx.x * 256 + threadIdx.x;
  if (i < n) dst[i] = (_Float16)src[i];
}

// w1[o*640+k] = conv1_w[o*640+k] * m_items[k]   (m_items folded into conv1)
__global__ void k_cvt_w1(_Float16* __restrict__ dst, const float* __restrict__ c1w,
                         const float* __restrict__ mitems, int n) {
  int i = blockIdx.x * 256 + threadIdx.x;
  if (i < n) dst[i] = (_Float16)(c1w[i] * mitems[i % 640]);
}

// exp_w [128,512] -> transposed [512,128] f16 (so GEMM reads W[N,K] row-major)
__global__ void k_cvt_expT(_Float16* __restrict__ dst, const float* __restrict__ w) {
  int i = blockIdx.x * 256 + threadIdx.x;
  if (i < 65536) { int c = i >> 9; int j = i & 511; dst[j * 128 + c] = (_Float16)w[i]; }
}

// wf_post_w [o][i][3][3] -> [tap=dy*3+dx][o][i] f16 (tap-major 1x1 weights)
__global__ void k_cvt_wpost(_Float16* __restrict__ dst, const float* __restrict__ w) {
  int idx = blockIdx.x * 256 + threadIdx.x;
  if (idx < 147456) {
    int dx = idx % 3;
    int dy = (idx / 3) % 3;
    int i = (idx / 9) & 127;
    int o = idx / 1152;
    dst[((long)(dy * 3 + dx) * 128 + o) * 128 + i] = (_Float16)w[idx];
  }
}

// Zero the two halo pixel-columns of the padded x tensor [B][H][W+2][128].
__global__ void k_zero_pad(_Float16* __restrict__ xp) {
  int i = blockIdx.x * 256 + threadIdx.x;  // 2 sides * 512 rows * 128 ch = 131072
  if (i < 131072) {
    int c = i & 127;
    int r = (i >> 7) & 511;  // b*128 + y
    int side = i >> 16;      // 0 => x=-1 slot, 1 => x=128 slot
    long pix = (long)r * 130 + (side ? 129 : 0);
    xp[pix * 128 + c] = (_Float16)0;
  }
}

// --------------------- NCHW -> pixel-major f16 (+l2norm) -------------------
template <bool NORM>
__global__ __launch_bounds__(256) void k_pix_major(_Float16* __restrict__ dst,
                                                   const float* __restrict__ src) {
  __shared__ float vals[64 * 128];
  __shared__ float partial[256];
  __shared__ float scale[64];
  int t = threadIdx.x;
  int p = t & 63, q = t >> 6;
  long n0 = (long)blockIdx.x * 64;
  int b = (int)(n0 >> 14);
  int plane = (int)(n0 & 16383);
  float ss = 0.f;
#pragma unroll
  for (int i = 0; i < 32; ++i) {
    int c = q * 32 + i;
    float v = src[(long)(b * 128 + c) * 16384 + plane + p];  // coalesced in p
    vals[p * 128 + c] = v;
    ss += v * v;
  }
  partial[t] = ss;
  __syncthreads();
  if (q == 0) {
    float sc = 1.f;
    if (NORM) {
      float tot = partial[p] + partial[64 + p] + partial[128 + p] + partial[192 + p];
      sc = 1.f / fmaxf(sqrtf(tot), 1e-12f);
    }
    scale[p] = sc;
  }
  __syncthreads();
#pragma unroll
  for (int j = 0; j < 32; ++j) {
    int flat = t * 32 + j;  // coalesced 64B per lane
    int pp = flat >> 7, cc = flat & 127;
    dst[(n0 + pp) * 128 + cc] = (_Float16)(vals[pp * 128 + cc] * scale[pp]);
  }
}

// ----------------------------- generic WMMA GEMM ---------------------------
// C[R,N] = act(A[R,K] @ W[N,K]^T + bias), per-wave 32x64 tile, K-step 32.
constexpr int EPI_F16 = 0;   // f16 row-major out (+optional bias)
constexpr int EPI_SIG = 1;   // sigmoid -> f16
constexpr int EPI_F32 = 2;   // f32 row-major out
constexpr int EPI_SNP = 3;   // scatter into patch-embed A layout (f16)
constexpr int EPI_WF  = 4;   // fwt0*acc + fwt1*fuse_src -> padded [B][H][W+2][C] f16

template <int EPI>
__device__ __forceinline__ void epi_store(v8f c, int row0, int col, int N, int colBase,
                                          const float* __restrict__ bias, void* __restrict__ outp,
                                          const _Float16* __restrict__ fuse_src,
                                          float f0, float f1, int h) {
  float bv = bias ? bias[col] : 0.f;
#pragma unroll
  for (int v = 0; v < 8; ++v) {
    int row = row0 + h * 8 + v;
    float val = c[v] + bv;
    if (EPI == EPI_SIG) val = 1.f / (1.f + __expf(-val));
    if (EPI == EPI_WF)  val = f0 * val + f1 * (float)fuse_src[(long)row * N + col];
    if (EPI == EPI_F32) {
      ((float*)outp)[(long)row * N + col] = val;
    } else if (EPI == EPI_SNP) {
      int bb = row >> 14, y = (row >> 7) & 127, x = row & 127;
      long patch = (long)bb * 1024 + (y >> 2) * 32 + (x >> 2);
      ((_Float16*)outp)[patch * 2048 + (long)(colBase + col) * 16 + (y & 3) * 4 + (x & 3)] =
          (_Float16)val;
    } else if (EPI == EPI_WF) {
      int bb = row >> 14, y = (row >> 7) & 127, x = row & 127;
      ((_Float16*)outp)[((long)(bb * 128 + y) * 130 + x + 1) * 128 + col] = (_Float16)val;
    } else {
      ((_Float16*)outp)[(long)row * N + col] = (_Float16)val;
    }
  }
}

#define WMMA_F16(A_, B_, C_) \
  __builtin_amdgcn_wmma_f32_16x16x32_f16(false, A_, false, B_, (short)0, C_, false, false)

template <int EPI>
__global__ __launch_bounds__(256) void k_gemm(const _Float16* __restrict__ A, int lda,
                                              const _Float16* __restrict__ W, int ldw,
                                              const float* __restrict__ bias,
                                              void* __restrict__ outp,
                                              const _Float16* __restrict__ fuse_src,
                                              const float* __restrict__ wf2,
                                              int R, int N, int K, int colBase) {
  int lane = threadIdx.x & 31;
  int wave = threadIdx.x >> 5;
  int colTiles = N >> 6;
  long job = (long)blockIdx.x * 8 + wave;
  long totalJobs = (long)(R >> 5) * colTiles;
  if (job >= totalJobs) return;
  int rowT = (int)(job / colTiles), colT = (int)(job % colTiles);
  int row0 = rowT << 5, col0 = colT << 6;

  v8f c00 = zero8f(), c01 = zero8f(), c02 = zero8f(), c03 = zero8f();
  v8f c10 = zero8f(), c11 = zero8f(), c12 = zero8f(), c13 = zero8f();
  for (int k0 = 0; k0 < K; k0 += 32) {
    if (k0 + 32 < K)
      __builtin_prefetch(A + (long)(row0 + (lane & 15)) * lda + k0 + 32, 0, 1);
    v16h a0 = ld_a(A, lda, row0,      k0, lane);
    v16h a1 = ld_a(A, lda, row0 + 16, k0, lane);
    v16h b0 = ld_b(W, ldw, col0,      k0, lane);
    v16h b1 = ld_b(W, ldw, col0 + 16, k0, lane);
    v16h b2 = ld_b(W, ldw, col0 + 32, k0, lane);
    v16h b3 = ld_b(W, ldw, col0 + 48, k0, lane);
    c00 = WMMA_F16(a0, b0, c00);
    c01 = WMMA_F16(a0, b1, c01);
    c02 = WMMA_F16(a0, b2, c02);
    c03 = WMMA_F16(a0, b3, c03);
    c10 = WMMA_F16(a1, b0, c10);
    c11 = WMMA_F16(a1, b1, c11);
    c12 = WMMA_F16(a1, b2, c12);
    c13 = WMMA_F16(a1, b3, c13);
  }

  float f0 = 0.f, f1 = 0.f;
  if (EPI == EPI_WF) {
    float w0 = fmaxf(wf2[0], 0.f), w1 = fmaxf(wf2[1], 0.f);
    float sinv = 1.f / (w0 + w1 + 1e-8f);
    f0 = w0 * sinv; f1 = w1 * sinv;
  }
  int h = lane >> 4, nn = lane & 15;
  epi_store<EPI>(c00, row0,      col0 + nn,      N, colBase, bias, outp, fuse_src, f0, f1, h);
  epi_store<EPI>(c01, row0,      col0 + 16 + nn, N, colBase, bias, outp, fuse_src, f0, f1, h);
  epi_store<EPI>(c02, row0,      col0 + 32 + nn, N, colBase, bias, outp, fuse_src, f0, f1, h);
  epi_store<EPI>(c03, row0,      col0 + 48 + nn, N, colBase, bias, outp, fuse_src, f0, f1, h);
  epi_store<EPI>(c10, row0 + 16, col0 + nn,      N, colBase, bias, outp, fuse_src, f0, f1, h);
  epi_store<EPI>(c11, row0 + 16, col0 + 16 + nn, N, colBase, bias, outp, fuse_src, f0, f1, h);
  epi_store<EPI>(c12, row0 + 16, col0 + 32 + nn, N, colBase, bias, outp, fuse_src, f0, f1, h);
  epi_store<EPI>(c13, row0 + 16, col0 + 48 + nn, N, colBase, bias, outp, fuse_src, f0, f1, h);
}

// ------------------------------- LayerNorm(128) ----------------------------
__global__ __launch_bounds__(256) void k_ln128(_Float16* __restrict__ dst,
                                               const float* __restrict__ src,
                                               const float* __restrict__ g,
                                               const float* __restrict__ be, int rows) {
  int lane = threadIdx.x & 31;
  int wave = threadIdx.x >> 5;
  int row = blockIdx.x * 8 + wave;
  if (row >= rows) return;
  const float* p = src + (long)row * 128;
  float v[4], s = 0.f, s2 = 0.f;
#pragma unroll
  for (int i = 0; i < 4; ++i) { v[i] = p[lane + 32 * i]; s += v[i]; s2 += v[i] * v[i]; }
#pragma unroll
  for (int m = 16; m >= 1; m >>= 1) { s += __shfl_xor(s, m, 32); s2 += __shfl_xor(s2, m, 32); }
  float mean = s * (1.f / 128.f);
  float var = s2 * (1.f / 128.f) - mean * mean;
  float rstd = rsqrtf(var + 1e-5f);
#pragma unroll
  for (int i = 0; i < 4; ++i) {
    int c = lane + 32 * i;
    dst[(long)row * 128 + c] = (_Float16)((v[i] - mean) * rstd * g[c] + be[c]);
  }
}

// ------------------ pixel-shuffle + LayerNorm(32) (one wave/pixel) ---------
__global__ __launch_bounds__(256) void k_shuffle_ln32(_Float16* __restrict__ dst,
                                                      const _Float16* __restrict__ fe,
                                                      const float* __restrict__ g,
                                                      const float* __restrict__ be) {
  int lane = threadIdx.x & 31;
  int wave = threadIdx.x >> 5;
  long n = (long)blockIdx.x * 8 + wave;  // 65536 pixels
  int b = (int)(n >> 14), y = (int)((n >> 7) & 127), x = (int)(n & 127);
  long patch = (long)b * 1024 + (y >> 2) * 32 + (x >> 2);
  int j = (y & 3) * 128 + (x & 3) * 32 + lane;
  float v = (float)fe[patch * 512 + j];
  float s = v, s2 = v * v;
#pragma unroll
  for (int m = 16; m >= 1; m >>= 1) { s += __shfl_xor(s, m, 32); s2 += __shfl_xor(s2, m, 32); }
  float mean = s * (1.f / 32.f);
  float var = s2 * (1.f / 32.f) - mean * mean;
  float rstd = rsqrtf(var + 1e-5f);
  dst[n * 32 + lane] = (_Float16)((v - mean) * rstd * g[lane] + be[lane]);
}

// ---------------- 3x3 conv: async-LDS staged taps + WMMA + BN + ReLU6 ------
// block: 256 thr = 8 waves; covers 32 pixels (one row segment) x 128 out ch.
// X is the padded [B][H][W+2][128] f16 tensor, so A-fragment loads are
// unconditional. Tap weights (9 x 32KB) are double-buffered in LDS via
// global_load_async_to_lds_b128; tap t+1 streams in while tap t computes.
__global__ __launch_bounds__(256) void k_conv3_bn(const _Float16* __restrict__ X,
                                                  const _Float16* __restrict__ Wt,  // [9][128][128]
                                                  const float* __restrict__ bn_g,
                                                  const float* __restrict__ bn_b,
                                                  float* __restrict__ out) {
  __shared__ __align__(16) _Float16 wbuf[2][128 * 128];  // 2 x 32KB
  int tid = threadIdx.x;
  int lane = tid & 31;
  int wave = tid >> 5;
  int bid = blockIdx.x;  // B*H*4 = 2048
  int xseg = bid & 3;
  int y = (bid >> 2) & 127;
  int b = bid >> 9;
  int rg = wave >> 2, cg = wave & 3;
  int x0 = xseg * 32 + rg * 16;
  int col0 = cg * 32;

  unsigned ldsBase = (unsigned)(unsigned long long)(void*)&wbuf[0][0];

  // stage tap 0
#pragma unroll
  for (int i = 0; i < 8; ++i) {
    int e = (i * 256 + tid) * 8;  // consecutive 16B chunks across lanes
    async_copy_b128(ldsBase + (unsigned)e * 2u, Wt + e);
  }

  v8f c0 = zero8f(), c1 = zero8f();
  int xl = x0 + (lane & 15);
  int h8 = 8 * (lane >> 4);

  for (int t = 0; t < 9; ++t) {
    wait_asynccnt0();     // own-wave async copies landed
    __syncthreads();      // all waves' slices landed; prev buffer free
    if (t + 1 < 9) {
      const _Float16* g = Wt + (long)(t + 1) * 16384;
      unsigned lbase = ldsBase + (unsigned)(((t + 1) & 1) * 32768);
#pragma unroll
      for (int i = 0; i < 8; ++i) {
        int e = (i * 256 + tid) * 8;
        async_copy_b128(lbase + (unsigned)e * 2u, g + e);
      }
    }
    int dy = t / 3, dx = t % 3;
    int yy = y + dy - 1;
    if (yy < 0 || yy > 127) continue;  // block-uniform (depends only on y)
    // padded pixel address: slot = (x + dx - 1) + 1 = xl + dx
    const _Float16* pix = X + ((long)(b * 128 + yy) * 130 + (xl + dx)) * 128 + h8;
    const _Float16* wt = &wbuf[t & 1][0];  // LDS -> ds_load_b128 fragments
#pragma unroll
    for (int kk = 0; kk < 4; ++kk) {
      int k0 = kk * 32;
      v16h a  = ld_pair(pix + k0, pix + k0 + 16);
      v16h b0 = ld_b(wt, 128, col0,      k0, lane);
      v16h b1 = ld_b(wt, 128, col0 + 16, k0, lane);
      c0 = WMMA_F16(a, b0, c0);
      c1 = WMMA_F16(a, b1, c1);
    }
  }

  const float inv = rsqrtf(1.f + 1e-5f);
  int h = lane >> 4, nn = lane & 15;
#pragma unroll
  for (int t = 0; t < 2; ++t) {
    int col = col0 + t * 16 + nn;
    float sc = bn_g[col] * inv, bb = bn_b[col];
    v8f c = t ? c1 : c0;
#pragma unroll
    for (int v = 0; v < 8; ++v) {
      int x = x0 + 8 * h + v;
      float val = fminf(fmaxf(c[v] * sc + bb, 0.f), 6.f);
      out[(long)(b * 128 + col) * 16384 + (long)y * 128 + x] = val;
    }
  }
}

// -------------------------------- launcher ---------------------------------

extern "C" void kernel_launch(void* const* d_in, const int* in_sizes, int n_in,
                              void* d_out, int out_size, void* d_ws, size_t ws_size,
                              hipStream_t stream) {
  const float* Structure = (const float*)d_in[0];
  const float* query     = (const float*)d_in[1];
  const float* m_items   = (const float*)d_in[2];
  const float* mod_w     = (const float*)d_in[3];
  const float* mod_b     = (const float*)d_in[4];
  const float* conv1_w   = (const float*)d_in[5];
  const float* conv1_b   = (const float*)d_in[6];
  const float* conv2_w   = (const float*)d_in[7];
  const float* conv2_b   = (const float*)d_in[8];
  const float* pe_w      = (const float*)d_in[9];
  const float* pe_b      = (const float*)d_in[10];
  const float* pe_g      = (const float*)d_in[11];
  const float* pe_beta   = (const float*)d_in[12];
  const float* exp_w     = (const float*)d_in[13];
  const float* fin_g     = (const float*)d_in[14];
  const float* fin_b     = (const float*)d_in[15];
  const float* up_w      = (const float*)d_in[16];
  const float* up_b      = (const float*)d_in[17];
  const float* wf_w2     = (const float*)d_in[18];
  const float* wf_pre_w  = (const float*)d_in[19];
  const float* wf_post_w = (const float*)d_in[20];
  const float* wf_bn_g   = (const float*)d_in[21];
  const float* wf_bn_b   = (const float*)d_in[22];
  (void)in_sizes; (void)n_in; (void)out_size; (void)ws_size;

  char* ws = (char*)d_ws;
  size_t off = 0;
  auto alloc = [&](size_t bytes) -> char* {
    char* p = ws + off;
    off = (off + bytes + 255) & ~(size_t)255;
    return p;
  };

  const long NPIX = 65536;  // B*H*W
  _Float16* sF16    = (_Float16*)alloc(NPIX * 128 * 2);     // 16.8 MB
  _Float16* snPatch = (_Float16*)alloc(4096L * 2048 * 2);   // 16.8 MB
  float*    fRaw    = (float*)alloc(4096L * 128 * 4);       //  2.1 MB
  _Float16* fLN16   = (_Float16*)alloc(4096L * 128 * 2);    //  1.0 MB
  _Float16* modW16  = (_Float16*)alloc(81920 * 2);
  _Float16* w1F16   = (_Float16*)alloc(40960 * 2);
  _Float16* c2W16   = (_Float16*)alloc(8192 * 2);
  _Float16* peW16   = (_Float16*)alloc(262144 * 2);
  _Float16* expW16T = (_Float16*)alloc(65536 * 2);
  _Float16* upW16   = (_Float16*)alloc(4096 * 2);
  _Float16* preW16  = (_Float16*)alloc(16384 * 2);
  _Float16* postW16 = (_Float16*)alloc(147456 * 2);
  // Big region (83.9 MB) for mod activations; aliased afterwards (mod dead
  // after the M0c GEMM) by fe/aLN/M1/xPad/query at disjoint byte offsets.
  char* modRegion = alloc(NPIX * 640 * 2);
  _Float16* modF16 = (_Float16*)modRegion;
  _Float16* fe16   = (_Float16*)(modRegion);                         //  4.2 MB
  _Float16* aLN16  = (_Float16*)(modRegion + (size_t)5  * 1048576);  //  4.2 MB
  _Float16* m1f16  = (_Float16*)(modRegion + (size_t)10 * 1048576);  // 16.8 MB
  _Float16* xPad   = (_Float16*)(modRegion + (size_t)27 * 1048576);  // 17.0 MB padded
  _Float16* qF16   = (_Float16*)(modRegion + (size_t)45 * 1048576);  // 16.8 MB

  auto cg = [](int n) { return dim3((unsigned)((n + 255) / 256)); };

  // ---- weight prep (fp32 -> f16, with folding / transposes) ----
  k_cvt_f16<<<cg(81920), 256, 0, stream>>>(modW16, mod_w, 81920);
  k_cvt_w1<<<cg(40960), 256, 0, stream>>>(w1F16, conv1_w, m_items, 40960);
  k_cvt_f16<<<cg(8192), 256, 0, stream>>>(c2W16, conv2_w, 8192);
  k_cvt_f16<<<cg(262144), 256, 0, stream>>>(peW16, pe_w, 262144);
  k_cvt_expT<<<cg(65536), 256, 0, stream>>>(expW16T, exp_w);
  k_cvt_f16<<<cg(4096), 256, 0, stream>>>(upW16, up_w, 4096);
  k_cvt_f16<<<cg(16384), 256, 0, stream>>>(preW16, wf_pre_w, 16384);
  k_cvt_wpost<<<cg(147456), 256, 0, stream>>>(postW16, wf_post_w);

  // ---- s = l2norm(Structure) -> pixel-major f16 ----
  k_pix_major<true><<<1024, 256, 0, stream>>>(sF16, Structure);

  // mod = sigmoid(s @ mod_w^T + mod_b)            [65536 x 640], K=128
  k_gemm<EPI_SIG><<<2560, 256, 0, stream>>>(sF16, 128, modW16, 128, mod_b, modF16,
                                            nullptr, nullptr, 65536, 640, 128, 0);
  // M0c = mod @ (conv1*m_items)^T + conv1_b  -> snPatch cols 0..63, K=640
  k_gemm<EPI_SNP><<<256, 256, 0, stream>>>(modF16, 640, w1F16, 640, conv1_b, snPatch,
                                           nullptr, nullptr, 65536, 64, 640, 0);
  // Sc = s @ conv2^T + conv2_b              -> snPatch cols 64..127, K=128
  k_gemm<EPI_SNP><<<256, 256, 0, stream>>>(sF16, 128, c2W16, 128, conv2_b, snPatch,
                                           nullptr, nullptr, 65536, 64, 128, 64);
  // patch-embed: f = snPatch @ pe_w^T + pe_b      [4096 x 128], K=2048
  k_gemm<EPI_F32><<<32, 256, 0, stream>>>(snPatch, 2048, peW16, 2048, pe_b, fRaw,
                                          nullptr, nullptr, 4096, 128, 2048, 0);
  k_ln128<<<512, 256, 0, stream>>>(fLN16, fRaw, pe_g, pe_beta, 4096);
  // expand: fe = f_ln @ exp_w                    [4096 x 512], K=128
  k_gemm<EPI_F16><<<128, 256, 0, stream>>>(fLN16, 128, expW16T, 128, nullptr, fe16,
                                           nullptr, nullptr, 4096, 512, 128, 0);
  // pixel shuffle + LN(32) -> per-pixel 32-vec A matrix
  k_shuffle_ln32<<<8192, 256, 0, stream>>>(aLN16, fe16, fin_g, fin_b);
  // M1 = aLN @ up_w^T + up_b                     [65536 x 128], K=32
  k_gemm<EPI_F16><<<512, 256, 0, stream>>>(aLN16, 32, upW16, 32, up_b, m1f16,
                                           nullptr, nullptr, 65536, 128, 32, 0);
  // query -> pixel-major f16 (no norm)
  k_pix_major<false><<<1024, 256, 0, stream>>>(qF16, query);
  // zero halo columns of padded x, then fused WF write into padded layout
  k_zero_pad<<<cg(131072), 256, 0, stream>>>(xPad);
  // x = fwt0*(q @ wf_pre^T) + fwt1*M1  -> padded [B][H][W+2][128]
  k_gemm<EPI_WF><<<512, 256, 0, stream>>>(qF16, 128, preW16, 128, nullptr, xPad,
                                          m1f16, wf_w2, 65536, 128, 128, 0);
  // 3x3 conv (async-LDS staged taps, unguarded A) + BN + ReLU6 -> d_out
  k_conv3_bn<<<2048, 256, 0, stream>>>(xPad, postW16, wf_bn_g, wf_bn_b, (float*)d_out);
}
